// TART_TCR_Block_68504728371657
// MI455X (gfx1250) — matrix operand order repaired and checked by reading, not compiled, not run on hardware
//
#include <hip/hip_runtime.h>
#include <hip/hip_bf16.h>
#include <stdint.h>

#define DM   1024    // d_model
#define DH   4096    // hidden 4*d_model
#define MP   50      // primes
#define MPAD 64      // padded primes (WMMA tiles)
#define BB   8       // batch
#define NN   4096    // tokens per batch

typedef __attribute__((ext_vector_type(16))) __bf16 v16bf;
typedef __attribute__((ext_vector_type(8)))  __bf16 v8bf;
typedef __attribute__((ext_vector_type(8)))  float  v8f;

// ---------- bf16 helpers (manual RNE conversion, storage-safe) ----------
__device__ __host__ inline __bf16 f2bf(float f) {
  union { float f; uint32_t u; } v; v.f = f;
  uint32_t r = (v.u + 0x7fffu + ((v.u >> 16) & 1u)) >> 16;
  unsigned short s = (unsigned short)r;
  __bf16 b;
  __builtin_memcpy(&b, &s, 2);
  return b;
}

// ---------- WMMA fragment loaders (CDNA5 16x16x32 bf16 layouts) ----------
// A matrix 16x32: lane L holds row L%16; element e -> k = (L>=16?8:0) + e%8 + 16*(e/8)
__device__ inline v16bf load_a(const __bf16* base, int ld) {
  int lane = threadIdx.x & 31;
  const __bf16* p = base + (lane & 15) * ld + ((lane & 16) ? 8 : 0);
  v8bf lo = *(const v8bf*)(p);
  v8bf hi = *(const v8bf*)(p + 16);
  v16bf a;
#pragma unroll
  for (int e = 0; e < 8; ++e) { a[e] = lo[e]; a[e + 8] = hi[e]; }
  return a;
}

// B matrix 32x16 loaded from row-major (col, k) storage ("B-transposed"):
// lane L holds column L%16, k = (L>=16?8:0) + e%8 + 16*(e/8)
__device__ inline v16bf load_bt(const __bf16* base, int ld) {
  int lane = threadIdx.x & 31;
  const __bf16* p = base + (lane & 15) * ld + ((lane & 16) ? 8 : 0);
  v8bf lo = *(const v8bf*)(p);
  v8bf hi = *(const v8bf*)(p + 16);
  v16bf b;
#pragma unroll
  for (int e = 0; e < 8; ++e) { b[e] = lo[e]; b[e + 8] = hi[e]; }
  return b;
}

__device__ inline v8f wmma_bf16(v16bf a, v16bf b, v8f c) {
  return __builtin_amdgcn_wmma_f32_16x16x32_bf16(
      /*neg_a=*/false, a, /*neg_b=*/false, b,
      /*c_mod=*/(short)0, c, /*reuse_a=*/false, /*reuse_b=*/false);
}

__device__ inline unsigned hashu(unsigned x) {
  x ^= x >> 16; x *= 0x7feb352dU;
  x ^= x >> 15; x *= 0x846ca68bU;
  x ^= x >> 16; return x;
}

// ---------- prep: normalize primes -> bf16, pad rows [50,64) with zeros ----------
__global__ void k_prep_primes(const float* __restrict__ primes, __bf16* __restrict__ p_hat) {
  int m = blockIdx.x;
  __shared__ float red[256];
  if (m >= MP) {
    for (int d = threadIdx.x; d < DM; d += 256) p_hat[(long)m * DM + d] = f2bf(0.f);
    return;
  }
  float ss = 0.f;
  for (int d = threadIdx.x; d < DM; d += 256) { float v = primes[(long)m * DM + d]; ss += v * v; }
  red[threadIdx.x] = ss; __syncthreads();
  for (int s = 128; s > 0; s >>= 1) { if (threadIdx.x < s) red[threadIdx.x] += red[threadIdx.x + s]; __syncthreads(); }
  float inv = 1.0f / fmaxf(sqrtf(red[0]), 1e-12f);
  for (int d = threadIdx.x; d < DM; d += 256)
    p_hat[(long)m * DM + d] = f2bf(primes[(long)m * DM + d] * inv);
}

// ---------- prep: per-token 1/||x|| and bf16 copy of x ----------
__global__ void k_prep_x(const float* __restrict__ x, __bf16* __restrict__ xbf,
                         float* __restrict__ inv_norm) {
  long t = blockIdx.x;
  const float* p = x + t * DM;
  __shared__ float red[256];
  float ss = 0.f;
  for (int d = threadIdx.x; d < DM; d += 256) { float v = p[d]; ss += v * v; }
  red[threadIdx.x] = ss; __syncthreads();
  for (int s = 128; s > 0; s >>= 1) { if (threadIdx.x < s) red[threadIdx.x] += red[threadIdx.x + s]; __syncthreads(); }
  if (threadIdx.x == 0) inv_norm[t] = 1.0f / fmaxf(sqrtf(red[0]), 1e-12f);
  for (int d = threadIdx.x; d < DM; d += 256) xbf[t * DM + d] = f2bf(p[d]);
}

// ---------- LDS-tiled transpose: src fp32 (R x C) -> dst bf16 (C x R) ----------
__global__ void k_transpose(const float* __restrict__ src, __bf16* __restrict__ dst,
                            int R, int C, long sbs, long dbs) {
  __shared__ __bf16 tile[32][33];
  const float* s = src + (long)blockIdx.z * sbs;
  __bf16* d = dst + (long)blockIdx.z * dbs;
  int c0 = blockIdx.x * 32, r0 = blockIdx.y * 32;
  for (int i = threadIdx.y; i < 32; i += 8)
    tile[i][threadIdx.x] = f2bf(s[(long)(r0 + i) * C + (c0 + threadIdx.x)]);
  __syncthreads();
  for (int i = threadIdx.y; i < 32; i += 8)
    d[(long)(c0 + i) * R + (r0 + threadIdx.x)] = tile[threadIdx.x][i];
}

// ---------- row sums of softmaxed bond matrix (== 1.0 analytically) ----------
__global__ void k_arowsum(const float* __restrict__ Ac, const float* __restrict__ Ah,
                          const float* __restrict__ Av, const float* wc, const float* wh,
                          const float* wv, float* __restrict__ scale) {
  int m = threadIdx.x;
  if (m >= MPAD) return;
  if (m >= MP) { scale[m] = 1.f; return; }
  float w0 = *wc, w1 = *wh, w2 = *wv;
  float row[MP];
  float mx = -1e30f;
  for (int n = 0; n < MP; ++n) {
    float v = w0 * Ac[m * MP + n] + w1 * Ah[m * MP + n] + w2 * Av[m * MP + n];
    row[n] = v; mx = fmaxf(mx, v);
  }
  float se = 0.f;
  for (int n = 0; n < MP; ++n) { row[n] = __expf(row[n] - mx); se += row[n]; }
  float s = 0.f;
  for (int n = 0; n < MP; ++n) s += row[n] / se;
  scale[m] = s;
}

// ---------- routing: logits via WMMA, softmax over M, store C both layouts ----------
__global__ void k_logits(const __bf16* __restrict__ xbf, const __bf16* __restrict__ p_hat,
                         const float* __restrict__ inv_norm,
                         __bf16* __restrict__ C_nm, __bf16* __restrict__ C_T) {
  int wid = threadIdx.x >> 5, lane = threadIdx.x & 31;
  int b = blockIdx.y;
  int t0 = (blockIdx.x * 4 + wid) * 16;
  const __bf16* xb = xbf + ((long)b * NN + t0) * DM;

  v8f acc[4];
#pragma unroll
  for (int j = 0; j < 4; ++j) acc[j] = (v8f){0.f, 0.f, 0.f, 0.f, 0.f, 0.f, 0.f, 0.f};

  for (int k = 0; k < DM; k += 32) {
    if (k + 256 < DM) __builtin_prefetch(xb + k + 256, 0, 1);
    v16bf a = load_a(xb + k, DM);
#pragma unroll
    for (int j = 0; j < 4; ++j) {
      v16bf bm = load_bt(p_hat + (long)(j * 16) * DM + k, DM);
      acc[j] = wmma_bf16(a, bm, acc[j]);
    }
  }

  int col = lane & 15;
  int rbase = (lane & 16) ? 8 : 0;
  const float* invp = inv_norm + (long)b * NN + t0 + rbase;
#pragma unroll
  for (int r = 0; r < 8; ++r) {
    float inv = invp[r] * 10.0f;  // /||x|| then /TEMP(0.1)
    float v[4];
#pragma unroll
    for (int j = 0; j < 4; ++j) {
      int m = j * 16 + col;
      v[j] = (m < MP) ? acc[j][r] * inv : -1e30f;  // mask padded primes
    }
    float mx = fmaxf(fmaxf(v[0], v[1]), fmaxf(v[2], v[3]));
#pragma unroll
    for (int s = 1; s < 16; s <<= 1) mx = fmaxf(mx, __shfl_xor(mx, s, 32));
    float e0 = __expf(v[0] - mx), e1 = __expf(v[1] - mx);
    float e2 = __expf(v[2] - mx), e3 = __expf(v[3] - mx);
    float se = e0 + e1 + e2 + e3;
#pragma unroll
    for (int s = 1; s < 16; s <<= 1) se += __shfl_xor(se, s, 32);
    float rinv = 1.0f / se;
    int token = t0 + rbase + r;
    float ev[4] = {e0, e1, e2, e3};
#pragma unroll
    for (int j = 0; j < 4; ++j) {
      int m = j * 16 + col;
      __bf16 cb = f2bf(ev[j] * rinv);
      C_nm[((long)b * NN + token) * MPAD + m] = cb;
      C_T[((long)b * MPAD + m) * NN + token] = cb;
    }
  }
}

// ---------- H_prime = C^T x  (per batch: 64x1024, K=4096), 16x64 strip/wave ----------
__global__ void k_hprime(const __bf16* __restrict__ C_T, const __bf16* __restrict__ xt,
                         const float* __restrict__ scale, float* __restrict__ Hp) {
  int wid = threadIdx.x >> 5, lane = threadIdx.x & 31;
  int b = blockIdx.y;
  int id = blockIdx.x * 4 + wid;          // 0..63
  int m0 = (id >> 4) * 16, d0 = (id & 15) * 64;
  const __bf16* A  = C_T + ((long)b * MPAD + m0) * NN;
  const __bf16* Bt = xt  + ((long)b * DM  + d0) * NN;
  v8f acc[4];
#pragma unroll
  for (int j = 0; j < 4; ++j) acc[j] = (v8f){0.f, 0.f, 0.f, 0.f, 0.f, 0.f, 0.f, 0.f};
  for (int k = 0; k < NN; k += 32) {
    if (k + 256 < NN) {
      __builtin_prefetch(A + k + 256, 0, 1);
      __builtin_prefetch(Bt + k + 256, 0, 1);
    }
    v16bf a = load_a(A + k, NN);
#pragma unroll
    for (int j = 0; j < 4; ++j)
      acc[j] = wmma_bf16(a, load_bt(Bt + (long)(j * 16) * NN + k, NN), acc[j]);
  }
  int col = lane & 15, rbase = (lane & 16) ? 8 : 0;
#pragma unroll
  for (int r = 0; r < 8; ++r) {
    int m = m0 + rbase + r;
    float sc = scale[m];
#pragma unroll
    for (int j = 0; j < 4; ++j)
      Hp[((long)b * MPAD + m) * DM + d0 + j * 16 + col] = acc[j][r] * sc;
  }
}

// ---------- polar factor via Newton-Schulz on 50x50 Gram + noise -> z (bf16) ----------
__global__ void k_polar(const float* __restrict__ Hp, __bf16* __restrict__ zbf) {
  __shared__ float Y[MP][MP + 2], Z[MP][MP + 2], T[MP][MP + 2];
  __shared__ float red[256];
  __shared__ float snorm;
  int b = blockIdx.x, tid = threadIdx.x;
  const float* H = Hp + (long)b * MPAD * DM;

  // G = H H^T  (50x50)
  for (int e = tid; e < MP * MP; e += 256) {
    int i = e / MP, j = e % MP;
    float s = 0.f;
    for (int d = 0; d < DM; ++d) s += H[(long)i * DM + d] * H[(long)j * DM + d];
    Y[i][j] = s;
  }
  __syncthreads();
  float ss = 0.f;
  for (int e = tid; e < MP * MP; e += 256) { float v = Y[e / MP][e % MP]; ss += v * v; }
  red[tid] = ss; __syncthreads();
  for (int s = 128; s > 0; s >>= 1) { if (tid < s) red[tid] += red[tid + s]; __syncthreads(); }
  if (tid == 0) snorm = sqrtf(red[0]) + 1e-20f;
  __syncthreads();
  float inv_s = 1.0f / snorm;
  for (int e = tid; e < MP * MP; e += 256) {
    int i = e / MP, j = e % MP;
    Y[i][j] *= inv_s;
    Z[i][j] = (i == j) ? 1.f : 0.f;
  }
  __syncthreads();

  // Coupled Newton-Schulz: Z -> Ghat^{-1/2}
  for (int it = 0; it < 25; ++it) {
    for (int e = tid; e < MP * MP; e += 256) {
      int i = e / MP, j = e % MP;
      float s = 0.f;
      for (int k = 0; k < MP; ++k) s += Z[i][k] * Y[k][j];
      T[i][j] = ((i == j) ? 1.5f : 0.f) - 0.5f * s;
    }
    __syncthreads();
    float ny[10], nz[10];
    int cnt = 0;
    for (int e = tid; e < MP * MP; e += 256) {
      int i = e / MP, j = e % MP;
      float sy = 0.f, sz = 0.f;
      for (int k = 0; k < MP; ++k) { sy += Y[i][k] * T[k][j]; sz += T[i][k] * Z[k][j]; }
      ny[cnt] = sy; nz[cnt] = sz; ++cnt;
    }
    __syncthreads();
    cnt = 0;
    for (int e = tid; e < MP * MP; e += 256) { Y[e / MP][e % MP] = ny[cnt]; Z[e / MP][e % MP] = nz[cnt]; ++cnt; }
    __syncthreads();
  }

  // H_logic = (1/sqrt(s)) Z H, plus sparse gaussian noise; store bf16 padded
  float zs = rsqrtf(snorm);
  for (long e = tid; e < (long)MP * DM; e += 256) {
    int i = (int)(e / DM), d = (int)(e % DM);
    float s = 0.f;
    for (int j = 0; j < MP; ++j) s += Z[i][j] * H[(long)j * DM + d];
    s *= zs;
    unsigned idx = (unsigned)(((b * MPAD + i) * DM) + d);
    unsigned u1 = hashu(idx * 2654435761u + 0x12345u);
    if ((u1 >> 8) * (1.0f / 16777216.0f) < 0.0464f) {
      unsigned u2 = hashu(idx * 2246822519u + 0x6789u);
      unsigned u3 = hashu(idx * 3266489917u + 0xabcdu);
      float r1 = ((u2 >> 8) + 0.5f) * (1.0f / 16777216.0f);
      float r2 = (u3 >> 8) * (1.0f / 16777216.0f);
      s += 0.1f * sqrtf(-2.f * __logf(r1)) * __cosf(6.2831853f * r2);
    }
    zbf[((long)b * MPAD + i) * DM + d] = f2bf(s);
  }
  for (long e = tid; e < (long)(MPAD - MP) * DM; e += 256) {
    int i = MP + (int)(e / DM), d = (int)(e % DM);
    zbf[((long)b * MPAD + i) * DM + d] = f2bf(0.f);
  }
}

// ---------- MLP layer 1: h = gelu(z W1 + b1), 16x64 strip per wave ----------
__global__ void k_mlp1(const __bf16* __restrict__ zbf, const __bf16* __restrict__ W1t,
                       const float* __restrict__ b1, __bf16* __restrict__ hbf) {
  int wid = threadIdx.x >> 5, lane = threadIdx.x & 31;
  int b = blockIdx.y;
  int id = blockIdx.x * 4 + wid;          // 0..255
  int m0 = (id >> 6) * 16, o0 = (id & 63) * 64;
  const __bf16* A  = zbf + ((long)b * MPAD + m0) * DM;
  const __bf16* Bt = W1t + (long)o0 * DM;
  int col = lane & 15, rbase = (lane & 16) ? 8 : 0;
  v8f acc[4];
#pragma unroll
  for (int j = 0; j < 4; ++j) {
    float bias = b1[o0 + j * 16 + col];
    acc[j] = (v8f){bias, bias, bias, bias, bias, bias, bias, bias};
  }
  for (int k = 0; k < DM; k += 32) {
    if (k + 256 < DM) {
      __builtin_prefetch(A + k + 256, 0, 1);
      __builtin_prefetch(Bt + k + 256, 0, 1);
    }
    v16bf a = load_a(A + k, DM);
#pragma unroll
    for (int j = 0; j < 4; ++j)
      acc[j] = wmma_bf16(a, load_bt(Bt + (long)(j * 16) * DM + k, DM), acc[j]);
  }
#pragma unroll
  for (int r = 0; r < 8; ++r) {
    long rowoff = ((long)b * MPAD + m0 + rbase + r) * DH;
#pragma unroll
    for (int j = 0; j < 4; ++j) {
      float v = acc[j][r];
      float g = 0.5f * v * (1.0f + erff(v * 0.70710678118f));  // exact GELU
      hbf[rowoff + o0 + j * 16 + col] = f2bf(g);
    }
  }
}

// ---------- MLP layer 2: H = tanh(h W2 + b2), stored transposed (d, m) ----------
__global__ void k_mlp2(const __bf16* __restrict__ hbf, const __bf16* __restrict__ W2t,
                       const float* __restrict__ b2, __bf16* __restrict__ Ht) {
  int wid = threadIdx.x >> 5, lane = threadIdx.x & 31;
  int b = blockIdx.y;
  int id = blockIdx.x * 4 + wid;          // 0..63
  int m0 = (id >> 4) * 16, d0 = (id & 15) * 64;
  const __bf16* A  = hbf + ((long)b * MPAD + m0) * DH;
  const __bf16* Bt = W2t + (long)d0 * DH;
  int col = lane & 15, rbase = (lane & 16) ? 8 : 0;
  v8f acc[4];
#pragma unroll
  for (int j = 0; j < 4; ++j) {
    float bias = b2[d0 + j * 16 + col];
    acc[j] = (v8f){bias, bias, bias, bias, bias, bias, bias, bias};
  }
  for (int k = 0; k < DH; k += 32) {
    if (k + 256 < DH) {
      __builtin_prefetch(A + k + 256, 0, 1);
      __builtin_prefetch(Bt + k + 256, 0, 1);
    }
    v16bf a = load_a(A + k, DH);
#pragma unroll
    for (int j = 0; j < 4; ++j)
      acc[j] = wmma_bf16(a, load_bt(Bt + (long)(j * 16) * DH + k, DH), acc[j]);
  }
#pragma unroll
  for (int j = 0; j < 4; ++j) {
    v8bf outv;
#pragma unroll
    for (int r = 0; r < 8; ++r) outv[r] = f2bf(tanhf(acc[j][r]));
    *(v8bf*)(Ht + ((long)b * DM + d0 + j * 16 + col) * MPAD + m0 + rbase) = outv;
  }
}

// ---------- decode: out = x + C H  (4096x1024, K=64), 16x64 strip per wave ----------
__global__ void k_final(const __bf16* __restrict__ C_nm, const __bf16* __restrict__ Ht,
                        const float* __restrict__ x, float* __restrict__ out) {
  int wid = threadIdx.x >> 5, lane = threadIdx.x & 31;
  int b = blockIdx.y;
  int id = blockIdx.x * 4 + wid;          // 0..4095
  int n0 = (id >> 4) * 16, d0 = (id & 15) * 64;
  const __bf16* A  = C_nm + ((long)b * NN + n0) * MPAD;
  const __bf16* Bt = Ht   + ((long)b * DM + d0) * MPAD;
  v8f acc[4];
#pragma unroll
  for (int j = 0; j < 4; ++j) acc[j] = (v8f){0.f, 0.f, 0.f, 0.f, 0.f, 0.f, 0.f, 0.f};
#pragma unroll
  for (int k = 0; k < 64; k += 32) {
    v16bf a = load_a(A + k, MPAD);
#pragma unroll
    for (int j = 0; j < 4; ++j)
      acc[j] = wmma_bf16(a, load_bt(Bt + (long)(j * 16) * MPAD + k, MPAD), acc[j]);
  }
  int col = lane & 15, rbase = (lane & 16) ? 8 : 0;
  const float* xp = x   + ((long)b * NN + n0 + rbase) * DM + d0 + col;
  float*       op = out + ((long)b * NN + n0 + rbase) * DM + d0 + col;
#pragma unroll
  for (int r = 0; r < 8; ++r) {
    long off = (long)r * DM;
#pragma unroll
    for (int j = 0; j < 4; ++j)
      op[off + j * 16] = acc[j][r] + xp[off + j * 16];
  }
}

extern "C" void kernel_launch(void* const* d_in, const int* in_sizes, int n_in,
                              void* d_out, int out_size, void* d_ws, size_t ws_size,
                              hipStream_t stream) {
  const float* x      = (const float*)d_in[0];
  const float* primes = (const float*)d_in[1];
  const float* wc     = (const float*)d_in[2];
  const float* wh     = (const float*)d_in[3];
  const float* wv     = (const float*)d_in[4];
  const float* Ac     = (const float*)d_in[5];
  const float* Ah     = (const float*)d_in[6];
  const float* Av     = (const float*)d_in[7];
  const float* W1     = (const float*)d_in[8];
  const float* b1     = (const float*)d_in[9];
  const float* W2     = (const float*)d_in[10];
  const float* b2     = (const float*)d_in[11];
  float* out = (float*)d_out;
  (void)in_sizes; (void)n_in; (void)out_size; (void)ws_size;

  char* ws = (char*)d_ws;
  size_t off = 0;
  auto alloc = [&](size_t bytes) -> void* {
    void* p = ws + off;
    off += (bytes + 255) & ~(size_t)255;
    return p;
  };
  __bf16* p_hat   = (__bf16*)alloc((size_t)MPAD * DM * 2);
  __bf16* xbf     = (__bf16*)alloc((size_t)BB * NN * DM * 2);
  __bf16* xt      = (__bf16*)alloc((size_t)BB * DM * NN * 2);
  float* inv_norm = (float*) alloc((size_t)BB * NN * 4);
  __bf16* C_nm    = (__bf16*)alloc((size_t)BB * NN * MPAD * 2);
  __bf16* C_T     = (__bf16*)alloc((size_t)BB * MPAD * NN * 2);
  float* scale    = (float*) alloc((size_t)MPAD * 4);
  float* Hp       = (float*) alloc((size_t)BB * MPAD * DM * 4);
  __bf16* zbf     = (__bf16*)alloc((size_t)BB * MPAD * DM * 2);
  __bf16* W1t     = (__bf16*)alloc((size_t)DM * DH * 2);
  __bf16* W2t     = (__bf16*)alloc((size_t)DM * DH * 2);
  __bf16* hbf     = (__bf16*)alloc((size_t)BB * MPAD * DH * 2);
  __bf16* Ht      = (__bf16*)alloc((size_t)BB * DM * MPAD * 2);

  dim3 tb(32, 8);
  k_prep_primes<<<MPAD, 256, 0, stream>>>(primes, p_hat);
  k_prep_x<<<BB * NN, 256, 0, stream>>>(x, xbf, inv_norm);
  k_transpose<<<dim3(DM / 32, NN / 32, BB), tb, 0, stream>>>(x, xt, NN, DM, (long)NN * DM, (long)DM * NN);
  k_transpose<<<dim3(DH / 32, DM / 32, 1), tb, 0, stream>>>(W1, W1t, DM, DH, 0, 0);
  k_transpose<<<dim3(DM / 32, DH / 32, 1), tb, 0, stream>>>(W2, W2t, DH, DM, 0, 0);
  k_arowsum<<<1, 64, 0, stream>>>(Ac, Ah, Av, wc, wh, wv, scale);
  k_logits<<<dim3(64, BB), 128, 0, stream>>>(xbf, p_hat, inv_norm, C_nm, C_T);
  k_hprime<<<dim3(16, BB), 128, 0, stream>>>(C_T, xt, scale, Hp);
  k_polar<<<BB, 256, 0, stream>>>(Hp, zbf);
  k_mlp1<<<dim3(64, BB), 128, 0, stream>>>(zbf, W1t, b1, hbf);
  k_mlp2<<<dim3(16, BB), 128, 0, stream>>>(hbf, W2t, b2, Ht);
  k_final<<<dim3(1024, BB), 128, 0, stream>>>(C_nm, Ht, x, out);
}
